// FeedzaiExtraProduction_53223234732113
// MI455X (gfx1250) — matrix operand order repaired and checked by reading, not compiled, not run on hardware
//
#include <hip/hip_runtime.h>
#include <hip/hip_bf16.h>

// ---------------- problem constants ----------------
#define B_TX   1000000
#define FEAT   64
#define UDIM   48
#define BANK   1000
#define G3     144            // 3*UDIM
#define WSTRIDE 72            // LDS stride (halves) for transposed weight rows (K padded 48->64, +8 pad)
#define HSTRIDE 52            // LDS stride (floats) for h_new rows (48 + 4 pad)
#define WAVES_PER_BLOCK 4
#define TILES_PER_WAVE  5
#define NBLOCKS 3125          // 3125 * 4 waves * 5 tiles * 16 rows = 1,000,000

typedef __attribute__((ext_vector_type(16))) __bf16 v16bf;
typedef __attribute__((ext_vector_type(8)))  float  v8f;

// ---------------- helpers ----------------
// Native f32 -> bf16 conversion (RNE on gfx1250); clang emits v_cvt_pk_bf16_f32.
static __device__ __forceinline__ __bf16 f2bf(float f) {
  return (__bf16)f;
}
static __device__ __forceinline__ unsigned short f2bfbits(float f) {
  __bf16 h = (__bf16)f;
  return __builtin_bit_cast(unsigned short, h);
}

struct U4x2 { uint4 a, b; };
// B-fragment: halves j<8 -> K=kb+j, j>=8 -> K=kb+16+(j-8) (lane n = N, kb = 0/8) -> 2 x b128 from LDS
static __device__ __forceinline__ v16bf ldfrag(const unsigned short* p) {
  U4x2 t;
  t.a = *(const uint4*)p;
  t.b = *(const uint4*)(p + 16);
  return __builtin_bit_cast(v16bf, t);
}

static __device__ __forceinline__ void put4(v16bf& v, int base, float4 f) {
  v[base + 0] = f2bf(f.x); v[base + 1] = f2bf(f.y);
  v[base + 2] = f2bf(f.z); v[base + 3] = f2bf(f.w);
}
// A-fragment (16x32 bf16, ISA layout): lane m=L%16, kb = (L<16)?0:8.
// halves j<8 -> K=kb+j ; j>=8 -> K=kb+16+(j-8)
static __device__ __forceinline__ v16bf afrag_full(const float* row, int kb) {
  v16bf v;
  put4(v, 0,  *(const float4*)(row + kb));
  put4(v, 4,  *(const float4*)(row + kb + 4));
  put4(v, 8,  *(const float4*)(row + kb + 16));
  put4(v, 12, *(const float4*)(row + kb + 20));
  return v;
}
// low half valid (K = kb..kb+7 relative), upper 8 halves zero (K >= 48 padding)
static __device__ __forceinline__ v16bf afrag_lowhalf(const float* row, int kb) {
  v16bf v;
  put4(v, 0, *(const float4*)(row + kb));
  put4(v, 4, *(const float4*)(row + kb + 4));
  const __bf16 z = __builtin_bit_cast(__bf16, (unsigned short)0);
#pragma unroll
  for (int j = 8; j < 16; ++j) v[j] = z;
  return v;
}

static __device__ __forceinline__ v8f wmma_bf16(v16bf a, v16bf b, v8f c) {
  return __builtin_amdgcn_wmma_f32_16x16x32_bf16(false, a, false, b, (short)0, c, false, false);
}
// first-K-step WMMA with inline-constant zero accumulator (no splat movs)
static __device__ __forceinline__ v8f wmma_bf16_z(v16bf a, v16bf b) {
  v8f z = {};
  return __builtin_amdgcn_wmma_f32_16x16x32_bf16(false, a, false, b, (short)0, z, false, false);
}
static __device__ __forceinline__ float sigmoidf(float x) { return 1.0f / (1.0f + __expf(-x)); }
// tanh via exp identity (clamped so e^{2x} stays finite); cheaper than ocml tanhf
static __device__ __forceinline__ float tanh_fast(float x) {
  float xc = fminf(fmaxf(x, -15.0f), 15.0f);
  float e = __expf(2.0f * xc);
  return (e - 1.0f) / (e + 1.0f);
}

// ---------------- kernel 0: init new_state copy + winners ----------------
__global__ void fz_init_kernel(const float* __restrict__ S, float* __restrict__ outState,
                               int* __restrict__ winners) {
  int i = blockIdx.x * 256 + threadIdx.x;
  if (i < BANK * UDIM) outState[i] = S[i];
  if (i < BANK) winners[i] = -1;
}

// ---------------- kernel 1: per-bank winning (max) transaction index ----------------
#define WBLK 512
#define WCHUNK 8192
__global__ __launch_bounds__(WBLK) void fz_winner_kernel(const int* __restrict__ card_ids,
                                                         int* __restrict__ winners) {
  __shared__ int lw[BANK];
  const int tid = threadIdx.x;
  for (int i = tid; i < BANK; i += WBLK) lw[i] = -1;
  __syncthreads();
  const int base = blockIdx.x * WCHUNK;
  for (int j = tid; j < WCHUNK; j += WBLK) {
    int i = base + j;
    if (i < B_TX) atomicMax(&lw[card_ids[i]], i);
  }
  __syncthreads();
  for (int i = tid; i < BANK; i += WBLK)
    if (lw[i] >= 0) atomicMax(&winners[i], lw[i]);
}

// ---------------- kernel 2: fused GRU + head + winner scatter ----------------
__global__ __launch_bounds__(128)
void fz_gru_main(const int*   __restrict__ card_ids,
                 const float* __restrict__ X,
                 const float* __restrict__ S,
                 const float* __restrict__ W,
                 const float* __restrict__ U,
                 const float* __restrict__ b,
                 const float* __restrict__ dw1,
                 const float* __restrict__ db1,
                 const float* __restrict__ dw2,
                 const float* __restrict__ db2,
                 const int*   __restrict__ winners,
                 float* __restrict__ out,        // [B]
                 float* __restrict__ outState) { // [BANK*UDIM]
  __shared__ unsigned short sW[G3 * WSTRIDE];   // W^T bf16: [n][k], k padded to 64
  __shared__ unsigned short sU[G3 * WSTRIDE];   // U^T bf16: [n][k], rows k>=48 zero
  __shared__ unsigned short sD[32 * WSTRIDE];   // dw1^T bf16: [n][k], n>=24 & k>=48 zero
  __shared__ float sB[G3];
  __shared__ float sDb1[32];
  __shared__ float sDw2[32];
  __shared__ float sDb2[1];
  __shared__ float sHn[WAVES_PER_BLOCK][16 * HSTRIDE];

  const int tid = threadIdx.x;
  // -------- cooperative weight staging (bf16, transposed, zero-padded) --------
  for (int idx = tid; idx < G3 * 64; idx += 128) {
    int n = idx >> 6, k = idx & 63;
    sW[n * WSTRIDE + k] = f2bfbits(W[k * G3 + n]);
    sU[n * WSTRIDE + k] = f2bfbits(k < UDIM ? U[k * G3 + n] : 0.0f);
  }
  for (int idx = tid; idx < 32 * 64; idx += 128) {
    int n = idx >> 6, k = idx & 63;
    sD[n * WSTRIDE + k] = f2bfbits((n < 24 && k < UDIM) ? dw1[k * 24 + n] : 0.0f);
  }
  for (int idx = tid; idx < G3; idx += 128) sB[idx] = b[idx];
  if (tid < 32) {
    sDb1[tid] = tid < 24 ? db1[tid] : 0.0f;
    sDw2[tid] = tid < 24 ? dw2[tid] : 0.0f;
  }
  if (tid == 0) sDb2[0] = db2[0];
  __syncthreads();

  const int lane   = tid & 31;
  const int wave   = tid >> 5;
  const int nl     = lane & 15;         // column within 16-wide tile (B/C/D lane = N)
  const int hi     = lane >> 4;
  const int kbA    = hi ? 8 : 0;        // A-fragment K base
  const int kbB    = hi ? 8 : 0;        // B-fragment K base
  const int rowoff = hi ? 8 : 0;        // C/D fragment row base
  const float db2v = sDb2[0];
  const float b1a  = sDb1[nl];
  const float b1b  = sDb1[16 + nl];
  const float w2a  = sDw2[nl];
  const float w2b  = sDw2[16 + nl];     // zero for cols >= 24
  float* sHnW = sHn[wave];

  const int wid = blockIdx.x * WAVES_PER_BLOCK + wave;

  for (int t = 0; t < TILES_PER_WAVE; ++t) {
    const int tile = wid * TILES_PER_WAVE + t;
    const int r0 = tile * 16;

    // ---- per-lane A-row (m = nl) ----
    const int rowA  = r0 + nl;
    const int cardA = card_ids[rowA];
    const float* xrow = X + (size_t)rowA * FEAT;
    const float* hrow = S + (size_t)cardA * UDIM;   // L2-resident gather

    v16bf ax0 = afrag_full(xrow, kbA);              // X k = 0..31
    v16bf ax1 = afrag_full(xrow + 32, kbA);         // X k = 32..63
    v16bf ah0 = afrag_full(hrow, kbA);              // H k = 0..31
    v16bf ah1 = afrag_lowhalf(hrow + 32, kbA);      // H k = 32..47, rest zero

    int cardC[8];                                    // cards for C-layout rows
#pragma unroll
    for (int i = 0; i < 8; ++i) cardC[i] = card_ids[r0 + rowoff + i];

    // ---- GRU gates, 3 column groups of 16 (covers all 48 units) ----
#pragma unroll
    for (int g = 0; g < 3; ++g) {
      const unsigned short* wz = &sW[((0 + g) * 16 + nl) * WSTRIDE + kbB];
      const unsigned short* wr = &sW[((3 + g) * 16 + nl) * WSTRIDE + kbB];
      const unsigned short* wh = &sW[((6 + g) * 16 + nl) * WSTRIDE + kbB];
      const unsigned short* uz = &sU[((0 + g) * 16 + nl) * WSTRIDE + kbB];
      const unsigned short* ur = &sU[((3 + g) * 16 + nl) * WSTRIDE + kbB];
      const unsigned short* uh = &sU[((6 + g) * 16 + nl) * WSTRIDE + kbB];

      const float bz = sB[(0 + g) * 16 + nl];
      const float br = sB[(3 + g) * 16 + nl];
      const float bh = sB[(6 + g) * 16 + nl];

      // all accumulators start from inline-constant zero (no splat movs)
      v8f xz = wmma_bf16_z(ax0, ldfrag(wz));
      xz = wmma_bf16(ax1, ldfrag(wz + 32), xz);
      v8f hz = wmma_bf16_z(ah0, ldfrag(uz));
      hz = wmma_bf16(ah1, ldfrag(uz + 32), hz);

      v8f xr = wmma_bf16_z(ax0, ldfrag(wr));
      xr = wmma_bf16(ax1, ldfrag(wr + 32), xr);
      v8f hr = wmma_bf16_z(ah0, ldfrag(ur));
      hr = wmma_bf16(ah1, ldfrag(ur + 32), hr);

      v8f xh = wmma_bf16_z(ax0, ldfrag(wh));
      xh = wmma_bf16(ax1, ldfrag(wh + 32), xh);
      v8f hh = wmma_bf16_z(ah0, ldfrag(uh));
      hh = wmma_bf16(ah1, ldfrag(uh + 32), hh);

      const int col = g * 16 + nl;
#pragma unroll
      for (int i = 0; i < 8; ++i) {
        float z  = sigmoidf(xz[i] + hz[i] + bz);     // bias folded into epilogue
        float r  = sigmoidf(xr[i] + hr[i] + br);
        float cd = tanh_fast(xh[i] + bh + r * hh[i]);
        float hv = S[(size_t)cardC[i] * UDIM + col];
        float hn = z * hv + (1.0f - z) * cd;
        sHnW[(rowoff + i) * HSTRIDE + col] = hn;     // stage in f32 C layout
      }
    }
    __syncthreads();   // uniform trip count across all 4 waves

    // ---- dense head: var = relu(h_new @ dw1 + db1); out = sigmoid(var @ dw2 + db2) ----
    const float* hn = sHnW + nl * HSTRIDE;
    v16bf an0 = afrag_full(hn, kbA);                 // h_new k = 0..31
    v16bf an1 = afrag_lowhalf(hn + 32, kbA);         // h_new k = 32..47

    const unsigned short* d0 = &sD[(0  + nl) * WSTRIDE + kbB];
    const unsigned short* d1 = &sD[(16 + nl) * WSTRIDE + kbB];
    v8f v0 = wmma_bf16_z(an0, ldfrag(d0));
    v0 = wmma_bf16(an1, ldfrag(d0 + 32), v0);
    v8f v1 = wmma_bf16_z(an0, ldfrag(d1));
    v1 = wmma_bf16(an1, ldfrag(d1 + 32), v1);

    float s[8];
#pragma unroll
    for (int i = 0; i < 8; ++i) {
      float a0 = fmaxf(v0[i] + b1a, 0.0f);           // bias folded into relu epilogue
      float a1 = fmaxf(v1[i] + b1b, 0.0f);
      s[i] = a0 * w2a + a1 * w2b;
    }
#pragma unroll
    for (int off = 1; off < 16; off <<= 1) {
#pragma unroll
      for (int i = 0; i < 8; ++i) s[i] += __shfl_xor(s[i], off, 32);
    }
    if (nl == 0) {
#pragma unroll
      for (int i = 0; i < 8; ++i)
        out[r0 + rowoff + i] = sigmoidf(s[i] + db2v);
    }

    // ---- last-write-wins scatter: only winning rows write the bank ----
    if (lane < 16) {
      if (winners[cardA] == rowA) {
        const float4* src = (const float4*)(sHnW + lane * HSTRIDE);
        float4* dst = (float4*)(outState + (size_t)cardA * UDIM);
#pragma unroll
        for (int q = 0; q < 12; ++q) dst[q] = src[q];
      }
    }
  }
}

// ---------------- launcher ----------------
extern "C" void kernel_launch(void* const* d_in, const int* in_sizes, int n_in,
                              void* d_out, int out_size, void* d_ws, size_t ws_size,
                              hipStream_t stream) {
  (void)in_sizes; (void)n_in; (void)out_size; (void)ws_size;
  const int*   card = (const int*)  d_in[0];
  const float* X    = (const float*)d_in[1];
  const float* S    = (const float*)d_in[2];
  const float* W    = (const float*)d_in[3];
  const float* U    = (const float*)d_in[4];
  const float* b    = (const float*)d_in[5];
  const float* dw1  = (const float*)d_in[6];
  const float* db1  = (const float*)d_in[7];
  const float* dw2  = (const float*)d_in[8];
  const float* db2  = (const float*)d_in[9];

  float* out      = (float*)d_out;          // [B] outputs
  float* outState = out + B_TX;             // [BANK*UDIM] new_state
  int*   winners  = (int*)d_ws;             // [BANK] winning tx index

  fz_init_kernel<<<(BANK * UDIM + 255) / 256, 256, 0, stream>>>(S, outState, winners);
  fz_winner_kernel<<<(B_TX + WCHUNK - 1) / WCHUNK, WBLK, 0, stream>>>(card, winners);
  fz_gru_main<<<NBLOCKS, 128, 0, stream>>>(card, X, S, W, U, b, dw1, db1, dw2, db2,
                                           winners, out, outState);
}
